// Block_56822417326088
// MI455X (gfx1250) — matrix operand order repaired and checked by reading, not compiled, not run on hardware
//
#include <hip/hip_runtime.h>
#include <math.h>

// ---------------------------------------------------------------------------
// Problem constants (fixed by the reference)
// ---------------------------------------------------------------------------
#define BB   8
#define TT   1024
#define DM   1024
#define NH   16
#define DH   64
#define SCALE 0.125f            // DH^-0.5
#define MROWS (BB*TT)           // 8192 rows for the projections
#define QSZ  ((size_t)BB*NH*TT*DH) // elements per Q / K / V plane (8,388,608)

typedef __attribute__((ext_vector_type(16))) __bf16 v16bf;
typedef __attribute__((ext_vector_type(8)))  float  v8f;

union BF16x16 { v16bf v; __bf16 b[16]; };

// A-matrix (16x32 bf16) per-lane K index for vgpr-pair p (0..7)
__device__ __forceinline__ int akk(int p, int half) {
  return half * 16 + 2 * p;
}

__device__ __forceinline__ v8f wmma_bf16(v16bf a, v16bf b, v8f c) {
  return __builtin_amdgcn_wmma_f32_16x16x32_bf16(false, a, false, b,
                                                 (short)0, c, false, false);
}

// ---------------------------------------------------------------------------
// Kernel 1: QKV projection.  qkv[r,e] = sum_d x[r,d] * w_qkv[e,d]
// Output written as bf16 into ws planes Q/K/V with layout [b][h][t][dh].
// Block: 256 threads = 8 waves; macro tile 128(M) x 64(N); K-step 64.
// ---------------------------------------------------------------------------
__global__ __launch_bounds__(256) void qkv_gemm_kernel(
    const float* __restrict__ X, const float* __restrict__ W,
    __bf16* __restrict__ qkv) {
  __shared__ __bf16 Al[128][72];   // 128x64 bf16 tile of X (+pad)
  __shared__ __bf16 Bl[64][72];    // 64x64  bf16 tile of W (+pad)

  const int tid  = threadIdx.x;
  const int lane = tid & 31, wv = tid >> 5;
  const int half = lane >> 4, ml = lane & 15;
  const int wm = (wv >> 1) * 32, wn = (wv & 1) * 32;
  const int m0 = blockIdx.y * 128, n0 = blockIdx.x * 64;

  const int am = tid >> 1,  ak = (tid & 1) * 32;   // A staging coords
  const int bn = tid >> 2,  bk = (tid & 3) * 16;   // B staging coords

  v8f acc[2][2];
  #pragma unroll
  for (int i = 0; i < 2; ++i)
    #pragma unroll
    for (int j = 0; j < 2; ++j) acc[i][j] = (v8f)(0.0f);

  for (int k0 = 0; k0 < DM; k0 += 64) {
    // --- stage A tile: 128x64 f32 -> bf16, 32 per thread (8x float4)
    {
      const float4* src = (const float4*)(X + (size_t)(m0 + am) * DM + k0 + ak);
      #pragma unroll
      for (int i = 0; i < 8; ++i) {
        float4 t = src[i];
        Al[am][ak + 4 * i + 0] = (__bf16)t.x;
        Al[am][ak + 4 * i + 1] = (__bf16)t.y;
        Al[am][ak + 4 * i + 2] = (__bf16)t.z;
        Al[am][ak + 4 * i + 3] = (__bf16)t.w;
      }
    }
    // --- stage B tile: 64x64 f32 -> bf16, 16 per thread (4x float4)
    {
      const float4* src = (const float4*)(W + (size_t)(n0 + bn) * DM + k0 + bk);
      #pragma unroll
      for (int i = 0; i < 4; ++i) {
        float4 t = src[i];
        Bl[bn][bk + 4 * i + 0] = (__bf16)t.x;
        Bl[bn][bk + 4 * i + 1] = (__bf16)t.y;
        Bl[bn][bk + 4 * i + 2] = (__bf16)t.z;
        Bl[bn][bk + 4 * i + 3] = (__bf16)t.w;
      }
    }
    // prefetch next K-tile while this one is consumed
    if (k0 + 64 < DM) {
      __builtin_prefetch(X + (size_t)(m0 + am) * DM + k0 + 64 + ak, 0, 0);
      __builtin_prefetch(W + (size_t)(n0 + bn) * DM + k0 + 64 + bk, 0, 0);
    }
    __syncthreads();

    BF16x16 af[2][2], bf[2][2];          // [tile][kstep]
    #pragma unroll
    for (int mi = 0; mi < 2; ++mi) {
      int row = wm + mi * 16 + ml;
      #pragma unroll
      for (int ks = 0; ks < 2; ++ks)
        #pragma unroll
        for (int p = 0; p < 8; ++p) {
          int kk = ks * 32 + akk(p, half);
          af[mi][ks].b[2 * p]     = Al[row][kk];
          af[mi][ks].b[2 * p + 1] = Al[row][kk + 1];
        }
    }
    #pragma unroll
    for (int ni = 0; ni < 2; ++ni) {
      int col = wn + ni * 16 + ml;
      #pragma unroll
      for (int ks = 0; ks < 2; ++ks)
        #pragma unroll
        for (int p = 0; p < 8; ++p) {
          int kk = ks * 32 + half * 16 + 2 * p;
          bf[ni][ks].b[2 * p]     = Bl[col][kk];
          bf[ni][ks].b[2 * p + 1] = Bl[col][kk + 1];
        }
    }
    #pragma unroll
    for (int ks = 0; ks < 2; ++ks)
      #pragma unroll
      for (int mi = 0; mi < 2; ++mi)
        #pragma unroll
        for (int ni = 0; ni < 2; ++ni)
          acc[mi][ni] = wmma_bf16(af[mi][ks].v, bf[ni][ks].v, acc[mi][ni]);
    __syncthreads();
  }

  // Epilogue: scatter bf16 into Q/K/V planes laid out [b][h][t][dh]
  #pragma unroll
  for (int mi = 0; mi < 2; ++mi)
    #pragma unroll
    for (int ni = 0; ni < 2; ++ni)
      #pragma unroll
      for (int r = 0; r < 8; ++r) {
        int m = m0 + wm + mi * 16 + half * 8 + r;    // global row (b*T+t)
        int n = n0 + wn + ni * 16 + ml;              // global col e in [0,3072)
        int b = m >> 10, t = m & 1023;
        int part = n >> 10, rem = n & 1023, h = rem >> 6, dh = rem & 63;
        size_t dst = (size_t)part * QSZ +
                     (((size_t)b * NH + h) * TT + t) * DH + dh;
        qkv[dst] = (__bf16)acc[mi][ni][r];
      }
}

// ---------------------------------------------------------------------------
// Kernel 2: fused causal attention with online softmax (flash style).
// Grid: (T/128, B*H); block 256 = 8 waves; each wave owns a 16-row Q tile,
// streams 32-wide K/V blocks, accumulates O (16x64) in fp32.
// ---------------------------------------------------------------------------
__global__ __launch_bounds__(256) void attn_kernel(
    const __bf16* __restrict__ qkv, __bf16* __restrict__ O) {
  __shared__ __bf16 Kl[32][72];     // K block [j][dh] (+pad)
  __shared__ __bf16 Vl[32][72];     // V block [j][dh]
  __shared__ __bf16 Pl[8][16][40];  // per-wave P scratch [m][j]

  const int tid  = threadIdx.x;
  const int lane = tid & 31, wv = tid >> 5;
  const int half = lane >> 4, ml = lane & 15;
  const int qb = blockIdx.x, bh = blockIdx.y;
  const int b = bh / NH, h = bh % NH;
  const int qrow0 = qb * 128 + wv * 16;

  const __bf16* Qp = qkv + (size_t)bh * TT * DH;
  const __bf16* Kp = qkv + QSZ + (size_t)bh * TT * DH;
  const __bf16* Vp = qkv + 2 * QSZ + (size_t)bh * TT * DH;

  // Q fragments (A-layout), two K-steps over DH=64
  BF16x16 qf[2];
  {
    const __bf16* qr = Qp + (size_t)(qrow0 + ml) * DH;
    #pragma unroll
    for (int ks = 0; ks < 2; ++ks)
      #pragma unroll
      for (int p = 0; p < 8; ++p) {
        int kk = ks * 32 + akk(p, half);
        qf[ks].b[2 * p]     = qr[kk];
        qf[ks].b[2 * p + 1] = qr[kk + 1];
      }
  }

  float rmax[8], rsum[8];
  v8f oacc[4];
  #pragma unroll
  for (int r = 0; r < 8; ++r) { rmax[r] = -3.0e38f; rsum[r] = 0.0f; }
  #pragma unroll
  for (int nt = 0; nt < 4; ++nt) oacc[nt] = (v8f)(0.0f);

  const int ldjj = tid >> 3, lddd = (tid & 7) * 8; // K/V staging coords
  const int jend = qb * 128 + 128;                 // causal upper bound
  for (int jb0 = 0; jb0 < jend; jb0 += 32) {
    __syncthreads();                               // protect previous readers
    {                                              // stage K/V: 8 bf16/thread
      const uint4* ks = (const uint4*)(Kp + (size_t)(jb0 + ldjj) * DH + lddd);
      const uint4* vs = (const uint4*)(Vp + (size_t)(jb0 + ldjj) * DH + lddd);
      uint4 kv = *ks, vv = *vs;
      unsigned int* kd = (unsigned int*)&Kl[ldjj][lddd];
      unsigned int* vd = (unsigned int*)&Vl[ldjj][lddd];
      kd[0] = kv.x; kd[1] = kv.y; kd[2] = kv.z; kd[3] = kv.w;
      vd[0] = vv.x; vd[1] = vv.y; vd[2] = vv.z; vd[3] = vv.w;
    }
    if (jb0 + 32 < jend) {
      __builtin_prefetch(Kp + (size_t)(jb0 + 32 + ldjj) * DH + lddd, 0, 0);
      __builtin_prefetch(Vp + (size_t)(jb0 + 32 + ldjj) * DH + lddd, 0, 0);
    }
    __syncthreads();

    if (jb0 <= qrow0 + 15) {                       // wave has live rows here
      // ---- S = Q K^T for two 16-col subtiles
      float sv[2][8];
      #pragma unroll
      for (int jn = 0; jn < 2; ++jn) {
        BF16x16 kb[2];
        #pragma unroll
        for (int ks = 0; ks < 2; ++ks)
          #pragma unroll
          for (int p = 0; p < 8; ++p) {
            int kk = ks * 32 + half * 16 + 2 * p;
            kb[ks].b[2 * p]     = Kl[jn * 16 + ml][kk];
            kb[ks].b[2 * p + 1] = Kl[jn * 16 + ml][kk + 1];
          }
        v8f s = (v8f)(0.0f);
        s = wmma_bf16(qf[0].v, kb[0].v, s);
        s = wmma_bf16(qf[1].v, kb[1].v, s);
        #pragma unroll
        for (int r = 0; r < 8; ++r) {
          int i = qrow0 + half * 8 + r;
          int j = jb0 + jn * 16 + ml;
          sv[jn][r] = (j <= i) ? s[r] * SCALE : -3.0e38f;
        }
      }
      // ---- online softmax: block row max
      float bmax[8];
      #pragma unroll
      for (int r = 0; r < 8; ++r) bmax[r] = fmaxf(sv[0][r], sv[1][r]);
      #pragma unroll
      for (int msk = 1; msk <= 8; msk <<= 1)
        #pragma unroll
        for (int r = 0; r < 8; ++r)
          bmax[r] = fmaxf(bmax[r], __shfl_xor(bmax[r], msk, 32));

      float corr[8], lsum[8], ps[2][8];
      #pragma unroll
      for (int r = 0; r < 8; ++r) {
        float nm = fmaxf(rmax[r], bmax[r]);
        corr[r]  = __expf(rmax[r] - nm);
        rmax[r]  = nm;
        lsum[r]  = 0.0f;
      }
      #pragma unroll
      for (int jn = 0; jn < 2; ++jn)
        #pragma unroll
        for (int r = 0; r < 8; ++r) {
          int i = qrow0 + half * 8 + r;
          int j = jb0 + jn * 16 + ml;
          float p = (j <= i) ? __expf(sv[jn][r] - rmax[r]) : 0.0f;
          ps[jn][r] = p;
          lsum[r] += p;
        }
      #pragma unroll
      for (int msk = 1; msk <= 8; msk <<= 1)
        #pragma unroll
        for (int r = 0; r < 8; ++r)
          lsum[r] += __shfl_xor(lsum[r], msk, 32);
      #pragma unroll
      for (int r = 0; r < 8; ++r) rsum[r] = rsum[r] * corr[r] + lsum[r];
      #pragma unroll
      for (int nt = 0; nt < 4; ++nt)
        #pragma unroll
        for (int r = 0; r < 8; ++r) oacc[nt][r] *= corr[r];

      // ---- relayout P (C-layout -> A-layout) through wave-local LDS
      #pragma unroll
      for (int jn = 0; jn < 2; ++jn)
        #pragma unroll
        for (int r = 0; r < 8; ++r)
          Pl[wv][half * 8 + r][jn * 16 + ml] = (__bf16)ps[jn][r];

      BF16x16 pa;
      #pragma unroll
      for (int p = 0; p < 8; ++p) {
        int kk = akk(p, half);
        pa.b[2 * p]     = Pl[wv][ml][kk];
        pa.b[2 * p + 1] = Pl[wv][ml][kk + 1];
      }
      // ---- O += P * V  (4 subtiles across dh)
      #pragma unroll
      for (int nt = 0; nt < 4; ++nt) {
        BF16x16 vb;
        #pragma unroll
        for (int p = 0; p < 8; ++p) {
          int kk = half * 16 + 2 * p;                // k == local j
          vb.b[2 * p]     = Vl[kk][nt * 16 + ml];
          vb.b[2 * p + 1] = Vl[kk + 1][nt * 16 + ml];
        }
        oacc[nt] = wmma_bf16(pa.v, vb.v, oacc[nt]);
      }
    }
  }

  // ---- normalize and store O as bf16, layout [b][t][h*64+dh]
  #pragma unroll
  for (int nt = 0; nt < 4; ++nt)
    #pragma unroll
    for (int r = 0; r < 8; ++r) {
      int t  = qrow0 + half * 8 + r;
      int dd = nt * 16 + ml;
      float val = oacc[nt][r] / rsum[r];
      O[((size_t)b * TT + t) * DM + h * DH + dd] = (__bf16)val;
    }
}

// ---------------------------------------------------------------------------
// Kernel 3: output projection.  out[r,e] = (sum_d O[r,d]*w_out[e,d] + b[e])*m[r]
// ---------------------------------------------------------------------------
__global__ __launch_bounds__(256) void out_gemm_kernel(
    const __bf16* __restrict__ Obf, const float* __restrict__ W,
    const float* __restrict__ bias, const float* __restrict__ msk,
    float* __restrict__ out) {
  __shared__ __bf16 Al[128][72];
  __shared__ __bf16 Bl[64][72];

  const int tid  = threadIdx.x;
  const int lane = tid & 31, wv = tid >> 5;
  const int half = lane >> 4, ml = lane & 15;
  const int wm = (wv >> 1) * 32, wn = (wv & 1) * 32;
  const int m0 = blockIdx.y * 128, n0 = blockIdx.x * 64;

  const int am = tid >> 1,  ak = (tid & 1) * 32;
  const int bn = tid >> 2,  bk = (tid & 3) * 16;

  v8f acc[2][2];
  #pragma unroll
  for (int i = 0; i < 2; ++i)
    #pragma unroll
    for (int j = 0; j < 2; ++j) acc[i][j] = (v8f)(0.0f);

  for (int k0 = 0; k0 < DM; k0 += 64) {
    {   // A already bf16: straight copy, 32 bf16 per thread (4x uint4)
      const uint4* src = (const uint4*)(Obf + (size_t)(m0 + am) * DM + k0 + ak);
      unsigned int* dst = (unsigned int*)&Al[am][ak];
      #pragma unroll
      for (int i = 0; i < 4; ++i) {
        uint4 t = src[i];
        dst[4 * i + 0] = t.x; dst[4 * i + 1] = t.y;
        dst[4 * i + 2] = t.z; dst[4 * i + 3] = t.w;
      }
    }
    {   // B from f32 w_out
      const float4* src = (const float4*)(W + (size_t)(n0 + bn) * DM + k0 + bk);
      #pragma unroll
      for (int i = 0; i < 4; ++i) {
        float4 t = src[i];
        Bl[bn][bk + 4 * i + 0] = (__bf16)t.x;
        Bl[bn][bk + 4 * i + 1] = (__bf16)t.y;
        Bl[bn][bk + 4 * i + 2] = (__bf16)t.z;
        Bl[bn][bk + 4 * i + 3] = (__bf16)t.w;
      }
    }
    if (k0 + 64 < DM) {
      __builtin_prefetch(Obf + (size_t)(m0 + am) * DM + k0 + 64 + ak, 0, 0);
      __builtin_prefetch(W + (size_t)(n0 + bn) * DM + k0 + 64 + bk, 0, 0);
    }
    __syncthreads();

    BF16x16 af[2][2], bf[2][2];
    #pragma unroll
    for (int mi = 0; mi < 2; ++mi) {
      int row = wm + mi * 16 + ml;
      #pragma unroll
      for (int ks = 0; ks < 2; ++ks)
        #pragma unroll
        for (int p = 0; p < 8; ++p) {
          int kk = ks * 32 + akk(p, half);
          af[mi][ks].b[2 * p]     = Al[row][kk];
          af[mi][ks].b[2 * p + 1] = Al[row][kk + 1];
        }
    }
    #pragma unroll
    for (int ni = 0; ni < 2; ++ni) {
      int col = wn + ni * 16 + ml;
      #pragma unroll
      for (int ks = 0; ks < 2; ++ks)
        #pragma unroll
        for (int p = 0; p < 8; ++p) {
          int kk = ks * 32 + half * 16 + 2 * p;
          bf[ni][ks].b[2 * p]     = Bl[col][kk];
          bf[ni][ks].b[2 * p + 1] = Bl[col][kk + 1];
        }
    }
    #pragma unroll
    for (int ks = 0; ks < 2; ++ks)
      #pragma unroll
      for (int mi = 0; mi < 2; ++mi)
        #pragma unroll
        for (int ni = 0; ni < 2; ++ni)
          acc[mi][ni] = wmma_bf16(af[mi][ks].v, bf[ni][ks].v, acc[mi][ni]);
    __syncthreads();
  }

  #pragma unroll
  for (int mi = 0; mi < 2; ++mi)
    #pragma unroll
    for (int ni = 0; ni < 2; ++ni)
      #pragma unroll
      for (int r = 0; r < 8; ++r) {
        int m = m0 + wm + mi * 16 + half * 8 + r;
        int n = n0 + wn + ni * 16 + ml;
        out[(size_t)m * DM + n] = (acc[mi][ni][r] + bias[n]) * msk[m];
      }
}

// ---------------------------------------------------------------------------
// Host-side launcher
// ---------------------------------------------------------------------------
extern "C" void kernel_launch(void* const* d_in, const int* in_sizes, int n_in,
                              void* d_out, int out_size, void* d_ws, size_t ws_size,
                              hipStream_t stream) {
  (void)in_sizes; (void)n_in; (void)out_size; (void)ws_size;
  const float* x     = (const float*)d_in[0];
  const float* m     = (const float*)d_in[1];
  const float* w_qkv = (const float*)d_in[2];
  const float* w_out = (const float*)d_in[3];
  const float* b_out = (const float*)d_in[4];
  float* out = (float*)d_out;

  __bf16* qkv_ws = (__bf16*)d_ws;            // 3*QSZ bf16 (48 MB)
  __bf16* o_ws   = qkv_ws + 3 * QSZ;         // B*T*DM bf16 (16 MB)

  // 1) QKV projection: M=8192, N=3072, K=1024
  qkv_gemm_kernel<<<dim3(3 * DM / 64, MROWS / 128), 256, 0, stream>>>(
      x, w_qkv, qkv_ws);
  // 2) fused causal attention
  attn_kernel<<<dim3(TT / 128, BB * NH), 256, 0, stream>>>(qkv_ws, o_ws);
  // 3) output projection + bias + padding mask: M=8192, N=1024, K=1024
  out_gemm_kernel<<<dim3(DM / 64, MROWS / 128), 256, 0, stream>>>(
      o_ws, w_out, b_out, m, out);
}
